// ContrastiveLoss_33715493274369
// MI455X (gfx1250) — matrix operand order repaired and checked by reading, not compiled, not run on hardware
//
#include <hip/hip_runtime.h>
#include <hip/hip_bf16.h>
#include <stdint.h>

typedef __attribute__((ext_vector_type(16))) __bf16 v16bf;
typedef __attribute__((ext_vector_type(8)))  float  v8f;

#define BATCH 16
#define CDIM  256
#define DDIM  4096
#define INV_T 2.0f          // 1/temperature, T = 0.5
#define EPSN  1e-12f

#define TILE  64            // output tile edge (64x64 sim tile per workgroup)
#define KB    64            // K-block staged in LDS per iteration
#define LDSK  72            // padded LDS row stride (uint16 units) -> conflict-free frag loads

// ---------- packed f32 -> 2 x bf16 (round-to-nearest-even), one VALU op ----------
__device__ __forceinline__ unsigned int pack2(float lo, float hi) {
  unsigned int r;
  asm("v_cvt_pk_bf16_f32 %0, %1, %2" : "=v"(r) : "v"(lo), "v"(hi));
  return r;
}

// =====================================================================
// Kernel 1: inverse channel norms  (normalize over dim=1, per (b, h*w))
// =====================================================================
__global__ __launch_bounds__(256) void chan_norm_kernel(
    const float* __restrict__ T, const float* __restrict__ S,
    float* __restrict__ tInv1, float* __restrict__ sInv1)
{
  const int pos = blockIdx.x * 256 + threadIdx.x;   // b*4096 + d
  const int b = pos >> 12;
  const int d = pos & (DDIM - 1);
  const float* tp = T + (size_t)b * CDIM * DDIM + d;
  const float* sp = S + (size_t)b * CDIM * DDIM + d;
  float st = 0.f, ss = 0.f;
  for (int c = 0; c < CDIM; ++c) {
    float x = tp[(size_t)c * DDIM];
    float y = sp[(size_t)c * DDIM];
    st = fmaf(x, x, st);
    ss = fmaf(y, y, ss);
  }
  tInv1[pos] = 1.0f / fmaxf(sqrtf(st), EPSN);
  sInv1[pos] = 1.0f / fmaxf(sqrtf(ss), EPSN);
}

// =====================================================================
// Kernel 2: inverse row norms of channel-normalized rows (per (b, c))
// =====================================================================
__global__ __launch_bounds__(256) void row_norm_kernel(
    const float* __restrict__ T, const float* __restrict__ S,
    const float* __restrict__ tInv1, const float* __restrict__ sInv1,
    float* __restrict__ tInvR, float* __restrict__ sInvR)
{
  __shared__ float red[16];
  const int row = blockIdx.x;          // b*256 + c
  const int b = row >> 8;
  const int tid = threadIdx.x;
  const int lane = tid & 31, wid = tid >> 5;
  const float* tp = T + (size_t)row * DDIM;
  const float* sp = S + (size_t)row * DDIM;
  const float* tc = tInv1 + b * DDIM;
  const float* sc = sInv1 + b * DDIM;
  float st = 0.f, ss = 0.f;
#pragma unroll
  for (int q = 0; q < 4; ++q) {
    int d = (q * 256 + tid) * 4;
    float4 x  = *(const float4*)&tp[d];
    float4 y  = *(const float4*)&sp[d];
    float4 ct = *(const float4*)&tc[d];
    float4 cs = *(const float4*)&sc[d];
    float a;
    a = x.x * ct.x; st = fmaf(a, a, st); a = x.y * ct.y; st = fmaf(a, a, st);
    a = x.z * ct.z; st = fmaf(a, a, st); a = x.w * ct.w; st = fmaf(a, a, st);
    a = y.x * cs.x; ss = fmaf(a, a, ss); a = y.y * cs.y; ss = fmaf(a, a, ss);
    a = y.z * cs.z; ss = fmaf(a, a, ss); a = y.w * cs.w; ss = fmaf(a, a, ss);
  }
#pragma unroll
  for (int off = 16; off > 0; off >>= 1) {
    st += __shfl_down(st, off, 32);
    ss += __shfl_down(ss, off, 32);
  }
  if (lane == 0) { red[wid] = st; red[8 + wid] = ss; }
  __syncthreads();
  if (tid == 0) {
    float a = 0.f, c2 = 0.f;
    for (int w = 0; w < 8; ++w) { a += red[w]; c2 += red[8 + w]; }
    tInvR[row] = 1.0f / fmaxf(sqrtf(a),  EPSN);
    sInvR[row] = 1.0f / fmaxf(sqrtf(c2), EPSN);
  }
}

// =====================================================================
// Kernel 3: batched GEMM sim = S_norm @ T_norm^T via bf16 WMMA,
//           fused with per-tile sum(exp(2*sim)) and diagonal extraction.
//           Double-buffered LDS with split fetch (before WMMA) / store
//           (after WMMA) so ds_store never sits in the WMMA wait window.
// =====================================================================
union AFrag { v16bf v; uint4 q[2]; };

struct TileRegs { float4 x[4]; float4 cv; };

// Phase 1: global -> registers (issued before the WMMA block; latency hides
// under the matrix ops via LOADcnt).
__device__ __forceinline__ void fetch_tile(
    const float* __restrict__ src, const float* __restrict__ cinv,
    int rowBase, int k0, int tid, TileRegs& tr)
{
  const int r0  = tid >> 4;           // 0..15
  const int col = (tid & 15) << 2;    // 0..60 step 4
  tr.cv = *(const float4*)&cinv[k0 + col];
#pragma unroll
  for (int rr = 0; rr < 4; ++rr) {
    int row = r0 + (rr << 4);
    const float* p = src + (size_t)(rowBase + row) * DDIM + k0 + col;
    tr.x[rr] = *(const float4*)p;
    // this call runs one block ahead already; prefetch covers block k0+2*KB
    if (k0 + KB < DDIM) __builtin_prefetch(p + KB, 0, 3);  // global_prefetch_b8
  }
}

// Phase 2: scale + cvt_pk_bf16 + ds_store (issued after the WMMA block).
__device__ __forceinline__ void store_tile(
    const TileRegs& tr, const float* __restrict__ rs4 /* 4 hoisted row scales */,
    unsigned short* lds, int tid)
{
  const int r0  = tid >> 4;
  const int col = (tid & 15) << 2;
#pragma unroll
  for (int rr = 0; rr < 4; ++rr) {
    int row = r0 + (rr << 4);
    float rs = rs4[rr];
    uint2 pk;
    pk.x = pack2(tr.x[rr].x * tr.cv.x * rs, tr.x[rr].y * tr.cv.y * rs);
    pk.y = pack2(tr.x[rr].z * tr.cv.z * rs, tr.x[rr].w * tr.cv.w * rs);
    *(uint2*)&lds[row * LDSK + col] = pk;
  }
}

__global__ __launch_bounds__(256) void gemm_lse_kernel(
    const float* __restrict__ S, const float* __restrict__ T,
    const float* __restrict__ sInv1, const float* __restrict__ tInv1,
    const float* __restrict__ sInvR, const float* __restrict__ tInvR,
    float2* __restrict__ partials)
{
  __shared__ unsigned short sA[2][TILE * LDSK];
  __shared__ unsigned short sB[2][TILE * LDSK];
  __shared__ float red[16];

  const int blk = blockIdx.x;               // 0..255
  const int b = blk >> 4;
  const int t = blk & 15;
  const int rowBase = (t >> 2) * TILE;      // student rows
  const int colBase = (t & 3) * TILE;       // teacher rows (sim cols)

  const float* Sb  = S + (size_t)b * CDIM * DDIM;
  const float* Tb  = T + (size_t)b * CDIM * DDIM;
  const float* scv = sInv1 + b * DDIM;
  const float* tcv = tInv1 + b * DDIM;
  const float* srv = sInvR + b * CDIM;
  const float* trv = tInvR + b * CDIM;

  const int tid  = threadIdx.x;
  const int lane = tid & 31;
  const int wid  = tid >> 5;
  const int waveR = wid >> 1;               // 0..3 -> 16-row slab
  const int waveC = wid & 1;                // 0..1 -> 32-col slab
  const int half  = lane >> 4;              // lane half (WMMA K split)
  const int lrow  = lane & 15;

  // Hoist k-invariant per-row inverse norms (each thread touches fixed rows).
  const int r0 = tid >> 4;
  float rsA[4], rsB[4];
#pragma unroll
  for (int rr = 0; rr < 4; ++rr) {
    rsA[rr] = srv[rowBase + r0 + (rr << 4)];
    rsB[rr] = trv[colBase + r0 + (rr << 4)];
  }

  v8f acc[2] = {};                          // two 16x16 f32 accumulators

  TileRegs ta, tb2;
  // prologue: fill buffer 0
  fetch_tile(Sb, scv, rowBase, 0, tid, ta);
  fetch_tile(Tb, tcv, colBase, 0, tid, tb2);
  store_tile(ta,  rsA, sA[0], tid);
  store_tile(tb2, rsB, sB[0], tid);

  int cur = 0;
  for (int k0 = 0; k0 < DDIM; k0 += KB) {
    __syncthreads();                        // buf[cur] ready; buf[cur^1] free
    const bool more = (k0 + KB < DDIM);
    if (more) {                             // issue global loads early
      fetch_tile(Sb, scv, rowBase, k0 + KB, tid, ta);
      fetch_tile(Tb, tcv, colBase, k0 + KB, tid, tb2);
    }
    const unsigned short* a_s = sA[cur];
    const unsigned short* b_s = sB[cur];
#pragma unroll
    for (int kk = 0; kk < KB; kk += 32) {
      AFrag af, bf0, bf1;
      {
        // A (16-bit 16x32): lanes 0-15 K=kk+0..7 / +16..23; lanes 16-31 K=+8..15 / +24..31
        int row = waveR * 16 + lrow;
        int kb  = kk + (half << 3);
        af.q[0] = *(const uint4*)&a_s[row * LDSK + kb];
        af.q[1] = *(const uint4*)&a_s[row * LDSK + kb + 16];
      }
      {
        // B (32x16): per-lane column = row of T; lanes 0-15 K=kk+0..7 / +16..23; lanes 16-31 K=+8..15 / +24..31
        int n0 = waveC * 32 + lrow;
        int kb = kk + (half << 3);
        bf0.q[0] = *(const uint4*)&b_s[n0 * LDSK + kb];
        bf0.q[1] = *(const uint4*)&b_s[n0 * LDSK + kb + 16];
        int n1 = n0 + 16;
        bf1.q[0] = *(const uint4*)&b_s[n1 * LDSK + kb];
        bf1.q[1] = *(const uint4*)&b_s[n1 * LDSK + kb + 16];
      }
      acc[0] = __builtin_amdgcn_wmma_f32_16x16x32_bf16(
          false, af.v, false, bf0.v, (short)0, acc[0], false, false);
      acc[1] = __builtin_amdgcn_wmma_f32_16x16x32_bf16(
          false, af.v, false, bf1.v, (short)0, acc[1], false, false);
    }
    if (more) {                             // convert + ds_store after compute
      store_tile(ta,  rsA, sA[cur ^ 1], tid);
      store_tile(tb2, rsB, sB[cur ^ 1], tid);
    }
    cur ^= 1;
  }

  // Epilogue: sim in [-1,1] so exp(2*sim) <= e^2; no max-shift needed for LSE.
  float se = 0.f, ps = 0.f;
  const int gr0 = rowBase + waveR * 16 + (half << 3);   // C/D layout: M = r + half*8
#pragma unroll
  for (int j = 0; j < 2; ++j) {
    const int gc = colBase + waveC * 32 + j * 16 + lrow;
#pragma unroll
    for (int r = 0; r < 8; ++r) {
      float v = acc[j][r];
      se += __expf(INV_T * v);
      if (gr0 + r == gc) ps += v;           // diagonal (positive pair)
    }
  }
#pragma unroll
  for (int off = 16; off > 0; off >>= 1) {
    se += __shfl_down(se, off, 32);
    ps += __shfl_down(ps, off, 32);
  }
  if (lane == 0) { red[wid] = se; red[8 + wid] = ps; }
  __syncthreads();
  if (tid == 0) {
    float a = 0.f, p = 0.f;
    for (int w = 0; w < 8; ++w) { a += red[w]; p += red[8 + w]; }
    partials[blk] = make_float2(a, p);      // fixed slot -> deterministic
  }
}

// =====================================================================
// Kernel 4: fold 16 tile partials per sample into the scalar loss
// =====================================================================
__global__ void finalize_kernel(const float2* __restrict__ partials,
                                float* __restrict__ out)
{
  if (threadIdx.x == 0 && blockIdx.x == 0) {
    float loss = 0.f;
    for (int b = 0; b < BATCH; ++b) {
      float se = 0.f, ps = 0.f;
      for (int t = 0; t < 16; ++t) {
        float2 p = partials[b * 16 + t];
        se += p.x; ps += p.y;
      }
      // per_sample = mean_c(-pos/T) + logsumexp(sim/T) = -2*sum(pos)/C + log(se)
      loss += logf(se) - (INV_T * ps) * (1.0f / (float)CDIM);
    }
    out[0] = loss * (1.0f / (float)BATCH);
  }
}

// =====================================================================
// Host-side launcher
// =====================================================================
extern "C" void kernel_launch(void* const* d_in, const int* in_sizes, int n_in,
                              void* d_out, int out_size, void* d_ws, size_t ws_size,
                              hipStream_t stream) {
  const float* teacher = (const float*)d_in[0];
  const float* student = (const float*)d_in[1];
  float* out = (float*)d_out;

  // Workspace layout (bytes):
  //   sInv1: [16][4096] f32 @ 0        (256 KB)
  //   tInv1: [16][4096] f32 @ 262144   (256 KB)
  //   sInvR: [16][256]  f32 @ 524288   (16 KB)
  //   tInvR: [16][256]  f32 @ 540672   (16 KB)
  //   partials: 256 x float2 @ 557056  (2 KB)
  char* ws = (char*)d_ws;
  float*  sInv1    = (float*)(ws + 0);
  float*  tInv1    = (float*)(ws + 262144);
  float*  sInvR    = (float*)(ws + 524288);
  float*  tInvR    = (float*)(ws + 540672);
  float2* partials = (float2*)(ws + 557056);

  chan_norm_kernel<<<256, 256, 0, stream>>>(teacher, student, tInv1, sInv1);
  row_norm_kernel<<<BATCH * CDIM, 256, 0, stream>>>(teacher, student,
                                                    tInv1, sInv1, tInvR, sInvR);
  gemm_lse_kernel<<<BATCH * 16, 256, 0, stream>>>(student, teacher,
                                                  sInv1, tInv1, sInvR, tInvR,
                                                  partials);
  finalize_kernel<<<1, 32, 0, stream>>>(partials, out);
}
